// GraphAttentionLayer_79328045957506
// MI455X (gfx1250) — matrix operand order repaired and checked by reading, not compiled, not run on hardware
//
#include <hip/hip_runtime.h>
#include <hip/hip_bf16.h>

// ---------------------------------------------------------------------------
// GAT layer for MI455X (gfx1250, wave32, WMMA).
//   B=4, N=2048, F=256, H=8, NH=32.
// Pipeline:
//   k0: WT = transpose(W) in bf16            (one-time, 128KB, makes k1's
//       B-operand loads contiguous 32B vectors)
//   k1: Wh = h @ W   via v_wmma_f32_16x16x32_bf16, stored transposed (bf16)
//       as WhT[b*8+h][nh][j]  (so k3's B-operand loads are contiguous)
//   k2: s1/s2 = Whh . a1/a2  (rank-1 score precompute)
//   k3: fused flash-attention: e = mask(lrelu(s1[i]+s2[j])), online softmax,
//       P(bf16) x V(bf16) via WMMA, f32 accum, ELU epilogue.
//       adj stream prefetched with global_prefetch.
// Workspace: WhT 4MB @0, s1 256KB @4MB, s2 256KB @4.25MB, WT 128KB @4.5MB.
// ---------------------------------------------------------------------------

typedef __attribute__((ext_vector_type(16))) __bf16 v16bf;
typedef __attribute__((ext_vector_type(8)))  float  v8f;
typedef __attribute__((ext_vector_type(4)))  float  f32x4;
typedef __attribute__((ext_vector_type(4)))  int    i32x4;

constexpr int kB  = 4;
constexpr int kN  = 2048;
constexpr int kF  = 256;   // IN_F == OUT_F
constexpr int kH  = 8;
constexpr int kNH = 32;

// ---------------------------------------------------------------------------
// Kernel 0: WT[n][k] = (bf16) W[k][n].  Coalesced reads, strided bf16 writes.
// ---------------------------------------------------------------------------
__global__ __launch_bounds__(256) void gat_transpose_w(const float* __restrict__ W,
                                                       __bf16* __restrict__ WT) {
  const int t = blockIdx.x * 256 + threadIdx.x;   // 65536
  const int k = t >> 8;
  const int n = t & 255;
  WT[(size_t)n * kF + k] = (__bf16)W[t];
}

// ---------------------------------------------------------------------------
// Kernel 1: Wh = h @ W  (M=8192, N=256, K=256), bf16 WMMA, f32 accumulate.
// grid.x = 512 (16-row tiles), 256 threads = 8 waves; wave w owns n-tile w*32.
// Output scattered into WhT[(b*8+hh)*32 + nh][j] (bf16).
// ---------------------------------------------------------------------------
__global__ __launch_bounds__(256) void gat_gemm_wh(const float* __restrict__ h,
                                                   const __bf16* __restrict__ WT,
                                                   __bf16* __restrict__ WhT) {
  const int lane = threadIdx.x & 31;
  const int wv   = threadIdx.x >> 5;     // n-tile within block
  const int n0   = wv * 32;
  const int m0   = blockIdx.x * 16;
  const int lr   = lane & 15;
  const bool hi  = lane >= 16;

  const float* hrow = h + (size_t)(m0 + lr) * kF;
  const int koff  = hi ? 16 : 0;         // A-operand K group for this half-wave
  const int col0  = n0 + lr;
  const int col1  = col0 + 16;
  // B-operand: lane n holds column N, K contiguous (0..15 lo-half, 16..31 hi-half)
  const __bf16* wt0 = WT + (size_t)col0 * kF + (hi ? 16 : 0);
  const __bf16* wt1 = WT + (size_t)col1 * kF + (hi ? 16 : 0);

  v8f c0 = {}; v8f c1 = {};
  for (int kc = 0; kc < kF; kc += 32) {
    // A (16x32 bf16): lane<16 holds K {0..15}, lane>=16 holds {16..31}
    f32x4 f0 = *(const f32x4*)(hrow + kc + koff);
    f32x4 f1 = *(const f32x4*)(hrow + kc + koff + 4);
    f32x4 f2 = *(const f32x4*)(hrow + kc + koff + 8);
    f32x4 f3 = *(const f32x4*)(hrow + kc + koff + 12);
    v16bf A;
#pragma unroll
    for (int e = 0; e < 4; ++e) {
      A[e]      = (__bf16)f0[e];
      A[e + 4]  = (__bf16)f1[e];
      A[e + 8]  = (__bf16)f2[e];
      A[e + 12] = (__bf16)f3[e];
    }
    const v16bf B0 = *(const v16bf*)(wt0 + kc);
    const v16bf B1 = *(const v16bf*)(wt1 + kc);
    c0 = __builtin_amdgcn_wmma_f32_16x16x32_bf16(false, A, false, B0, (short)0, c0, false, false);
    c1 = __builtin_amdgcn_wmma_f32_16x16x32_bf16(false, A, false, B1, (short)0, c1, false, false);
  }

  // Scatter into transposed per-head layout. Raw-view mapping:
  //   element (m, f):  b=m>>11, nb=m&2047, hh=nb>>8, j=(nb&255)*8 + (f>>5), nh=f&31
#pragma unroll
  for (int r = 0; r < 8; ++r) {
    const int R  = r + (hi ? 8 : 0);
    const int m  = m0 + R;
    const int bb = m >> 11;
    const int nb = m & 2047;
    const int hh = nb >> 8;
    const size_t baseBH = (size_t)((bb * kH + hh) * kNH);
    const int jj = (nb & 255) * 8;
    WhT[(baseBH + (col0 & 31)) * kN + jj + (col0 >> 5)] = (__bf16)c0[r];
    WhT[(baseBH + (col1 & 31)) * kN + jj + (col1 >> 5)] = (__bf16)c1[r];
  }
}

// ---------------------------------------------------------------------------
// Kernel 2: s1[t]=Whh[j].a1, s2[t]=Whh[j].a2  for t = (b*8+hh)*2048 + j.
// WhT column reads are coalesced across consecutive t.
// ---------------------------------------------------------------------------
__global__ __launch_bounds__(256) void gat_scores(const __bf16* __restrict__ WhT,
                                                  const float* __restrict__ a,
                                                  float* __restrict__ s1,
                                                  float* __restrict__ s2) {
  const int t   = blockIdx.x * 256 + threadIdx.x;   // 65536 total
  const int bhh = t >> 11;
  const int j   = t & 2047;
  const __bf16* base = WhT + (size_t)bhh * kNH * kN + j;
  float d1 = 0.f, d2 = 0.f;
#pragma unroll
  for (int k = 0; k < kNH; ++k) {
    const float v = (float)base[(size_t)k * kN];
    d1 += v * a[k];
    d2 += v * a[kNH + k];
  }
  s1[t] = d1;
  s2[t] = d2;
}

// ---------------------------------------------------------------------------
// Kernel 3: fused flash attention + ELU.
// One wave per (b, head, 16-row tile): 4096 waves = 512 blocks x 8 waves.
// ---------------------------------------------------------------------------
__global__ __launch_bounds__(256) void gat_attn(const int* __restrict__ adj,
                                                const __bf16* __restrict__ WhT,
                                                const float* __restrict__ s1,
                                                const float* __restrict__ s2,
                                                float* __restrict__ out) {
  const int lane = threadIdx.x & 31;
  const int wt   = blockIdx.x * 8 + (threadIdx.x >> 5);  // 0..4095
  const int bhh  = wt >> 7;                              // b*8+hh
  const int i0   = (wt & 127) * 16;
  const int b    = bhh >> 3;
  const int lr   = lane & 15;
  const bool hi  = lane >= 16;
  const int cb   = hi ? 16 : 0;     // j-column group of this half-wave (A layout)

  const float s1r        = s1[bhh * kN + i0 + lr];     // this lane's P row
  const int*   adjrow    = adj + ((size_t)b * kN + (i0 + lr)) * kN;
  const float* s2row     = s2 + bhh * kN;
  const __bf16* Vt       = WhT + (size_t)bhh * kNH * kN;
  const __bf16* vp0      = Vt + (size_t)lr * kN        + (hi ? 16 : 0);
  const __bf16* vp1      = Vt + (size_t)(lr + 16) * kN + (hi ? 16 : 0);
  const int rsel = hi ? 8 : 0;

  float m = -3.0e38f, l = 0.f;
  v8f acc0 = {}; v8f acc1 = {};

  for (int jc = 0; jc < kN; jc += 32) {
    // Prefetch the adj stream one chunk-group ahead (global_prefetch_b8).
    if (jc + 512 < kN) __builtin_prefetch(adjrow + jc + 512, 0, 1);

    // This lane's 16 j-columns: jc+cb+{0..15}
    const i32x4 a0 = *(const i32x4*)(adjrow + jc + cb);
    const i32x4 a1 = *(const i32x4*)(adjrow + jc + cb + 4);
    const i32x4 a2 = *(const i32x4*)(adjrow + jc + cb + 8);
    const i32x4 a3 = *(const i32x4*)(adjrow + jc + cb + 12);
    const f32x4 q0 = *(const f32x4*)(s2row + jc + cb);
    const f32x4 q1 = *(const f32x4*)(s2row + jc + cb + 4);
    const f32x4 q2 = *(const f32x4*)(s2row + jc + cb + 8);
    const f32x4 q3 = *(const f32x4*)(s2row + jc + cb + 12);

    float ev[16];
#pragma unroll
    for (int e = 0; e < 4; ++e) {
      float t0 = s1r + q0[e]; t0 = t0 >= 0.f ? t0 : 0.2f * t0; ev[e]      = (a0[e] > 0) ? t0 : -9.0e15f;
      float t1 = s1r + q1[e]; t1 = t1 >= 0.f ? t1 : 0.2f * t1; ev[e + 4]  = (a1[e] > 0) ? t1 : -9.0e15f;
      float t2 = s1r + q2[e]; t2 = t2 >= 0.f ? t2 : 0.2f * t2; ev[e + 8]  = (a2[e] > 0) ? t2 : -9.0e15f;
      float t3 = s1r + q3[e]; t3 = t3 >= 0.f ? t3 : 0.2f * t3; ev[e + 12] = (a3[e] > 0) ? t3 : -9.0e15f;
    }

    // Online softmax: row data lives in lanes L and L+16.
    float cmax = ev[0];
#pragma unroll
    for (int e = 1; e < 16; ++e) cmax = fmaxf(cmax, ev[e]);
    cmax = fmaxf(cmax, __shfl_xor(cmax, 16));
    const float mnew = fmaxf(m, cmax);
    const float corr = __expf(m - mnew);

    float psum = 0.f;
    v16bf A;
#pragma unroll
    for (int e = 0; e < 16; ++e) {
      const float p = __expf(ev[e] - mnew);
      psum += p;
      A[e] = (__bf16)p;
    }
    psum += __shfl_xor(psum, 16);
    l = l * corr + psum;
    m = mnew;

    // Rescale accumulator: acc element r belongs to row r+rsel, whose corr is
    // held (identically) by lanes (r+rsel) and (r+rsel+16).
#pragma unroll
    for (int r = 0; r < 8; ++r) {
      const float cr = __shfl(corr, r + rsel);
      acc0[r] *= cr;
      acc1[r] *= cr;
    }

    // B operands: 16 contiguous bf16 from the pre-transposed V (32B aligned).
    const v16bf B0 = *(const v16bf*)(vp0 + jc);
    const v16bf B1 = *(const v16bf*)(vp1 + jc);
    acc0 = __builtin_amdgcn_wmma_f32_16x16x32_bf16(false, A, false, B0, (short)0, acc0, false, false);
    acc1 = __builtin_amdgcn_wmma_f32_16x16x32_bf16(false, A, false, B1, (short)0, acc1, false, false);
  }

  // Epilogue: 1/l normalize, ELU, store. Output raw-view layout:
  //   out_flat = bhh*65536 + i*32 + nh
  float* obase = out + (size_t)bhh * (kN * kNH);
#pragma unroll
  for (int r = 0; r < 8; ++r) {
    const float lrow = __shfl(l, r + rsel);
    const float inv  = 1.f / lrow;
    const int irow   = i0 + r + rsel;
    float o0 = acc0[r] * inv;
    float o1 = acc1[r] * inv;
    o0 = o0 > 0.f ? o0 : (__expf(o0) - 1.f);
    o1 = o1 > 0.f ? o1 : (__expf(o1) - 1.f);
    obase[(size_t)irow * kNH + lr]      = o0;
    obase[(size_t)irow * kNH + 16 + lr] = o1;
  }
}

// ---------------------------------------------------------------------------
extern "C" void kernel_launch(void* const* d_in, const int* in_sizes, int n_in,
                              void* d_out, int out_size, void* d_ws, size_t ws_size,
                              hipStream_t stream) {
  const float* h   = (const float*)d_in[0];   // [4,2048,256] f32
  const int*   adj = (const int*)  d_in[1];   // [4,2048,2048] i32
  const float* W   = (const float*)d_in[2];   // [256,256] f32
  const float* a   = (const float*)d_in[3];   // [64,1] f32
  float*       out = (float*)d_out;           // [4,2048,256] f32

  // Workspace carving (needs 4.625 MB).
  __bf16* WhT = (__bf16*)d_ws;                                            // 4 MB
  float*  s1  = (float*)((char*)d_ws + (size_t)4 * 1024 * 1024);          // 256 KB
  float*  s2  = s1 + kB * kH * kN;                                        // 256 KB
  __bf16* WT  = (__bf16*)((char*)d_ws + (size_t)4 * 1024 * 1024 + 512 * 1024); // 128 KB

  gat_transpose_w<<<256, 256, 0, stream>>>(W, WT);
  gat_gemm_wh    <<<512, 256, 0, stream>>>(h, WT, WhT);
  gat_scores     <<<256, 256, 0, stream>>>(WhT, a, s1, s2);
  gat_attn       <<<512, 256, 0, stream>>>(adj, WhT, s1, s2, out);
}
